// XEyTransformerLayer_33122787787641
// MI455X (gfx1250) — matrix operand order repaired and checked by reading, not compiled, not run on hardware
//
#include <hip/hip_runtime.h>
#include <hip/hip_bf16.h>

// ---------------- constants ----------------
constexpr int cBS = 2, cN = 384, cDX = 256, cDE = 64, cDY = 128, cDT = 256;
constexpr int cFFX = 1024, cFFE = 128, cFFY = 512;
constexpr float cEPS = 1e-5f;

typedef __attribute__((ext_vector_type(16))) _Float16 v16h;
typedef __attribute__((ext_vector_type(8)))  float    v8f;
typedef int v4i_ __attribute__((vector_size(16)));

// ---- CDNA5 async global->LDS copy (guarded; falls back to plain loads) ----
#if defined(__gfx1250__) && __has_builtin(__builtin_amdgcn_global_load_async_to_lds_b128)
#define HAS_ASYNC_LDS 1
#if __has_builtin(__builtin_amdgcn_s_wait_asynccnt)
#define WAIT_ASYNC() __builtin_amdgcn_s_wait_asynccnt(0)
#else
#define WAIT_ASYNC() asm volatile("s_wait_asynccnt 0" ::: "memory")
#endif
#define ASYNC_B128(gsrc, ldst)                                              \
  __builtin_amdgcn_global_load_async_to_lds_b128(                           \
      (__attribute__((address_space(1))) v4i_*)(gsrc),                      \
      (__attribute__((address_space(3))) v4i_*)(ldst), 0, 0)
#else
#define HAS_ASYNC_LDS 0
#define WAIT_ASYNC() do {} while (0)
#endif

// ---------------- kernel parameter block ----------------
struct KP {
  // inputs
  const float* __restrict__ X; const float* __restrict__ E;
  const float* __restrict__ Y; const float* __restrict__ T;
  const int*   __restrict__ M;
  // fp32 biases / small weights
  const float * __restrict__ qB, * __restrict__ kB, * __restrict__ vB,
              * __restrict__ emB, * __restrict__ eaB, * __restrict__ xoB, * __restrict__ eoB;
  const float * __restrict__ lX1B, * __restrict__ lX2B, * __restrict__ lE1B, * __restrict__ lE2B;
  const float * __restrict__ ly1W, * __restrict__ ly1B, * __restrict__ ly2W, * __restrict__ ly2B;
  const float * __restrict__ nX1g, * __restrict__ nX1b, * __restrict__ nX2g, * __restrict__ nX2b,
              * __restrict__ nE1g, * __restrict__ nE1b, * __restrict__ nE2g, * __restrict__ nE2b,
              * __restrict__ ny1g, * __restrict__ ny1b, * __restrict__ ny2g, * __restrict__ ny2b;
  // f16 weights, pre-packed into WMMA B-fragment order (workspace)
  const _Float16 * __restrict__ WqH, * __restrict__ WkH, * __restrict__ WvH, * __restrict__ WxoH,
                 * __restrict__ WemH, * __restrict__ WeaH, * __restrict__ WeoH,
                 * __restrict__ WE1H, * __restrict__ WE2H, * __restrict__ WX1H, * __restrict__ WX2H;
  // film embeddings (workspace, per batch: [scale | shift])
  const float * __restrict__ fX1, * __restrict__ fX2, * __restrict__ fE1, * __restrict__ fE2;
  // activation buffers (workspace)
  float * __restrict__ Qb, * __restrict__ Kb, * __restrict__ Vb, * __restrict__ wVb;
  // outputs
  float * __restrict__ Xout, * __restrict__ Eout, * __restrict__ yout;
};

// ---------------- WMMA helpers (wave32 layouts per CDNA5 ISA) ----------------
__device__ __forceinline__ v8f wmma_f16(v16h a, v16h b, v8f c) {
  return __builtin_amdgcn_wmma_f32_16x16x32_f16(false, a, false, b, (short)0, c, false, false);
}

// A fragment: 16x32 f16 tile, row-major, leading dim lda.
__device__ __forceinline__ v16h ldA(const _Float16* A, int lda) {
  int lane = threadIdx.x & 31;
  int row  = lane & 15;
  int kb   = (lane < 16) ? 0 : 8;
  v16h a;
#pragma unroll
  for (int v = 0; v < 8; ++v) {
    int kk = kb + ((v < 4) ? (2*v) : (16 + 2*(v-4)));
    a[2*v]   = A[row*lda + kk];
    a[2*v+1] = A[row*lda + kk + 1];
  }
  return a;
}

// B fragment from pre-packed weights: one aligned 32B load per lane.
// Packed layout: P[((nt*ktiles + kt)*32 + lane)*16 + e]
__device__ __forceinline__ v16h ldBp(const _Float16* base, int ktiles, int nt, int kt) {
  int lane = threadIdx.x & 31;
  const _Float16* P = base + ((size_t)(nt * ktiles + kt) * 32 + lane) * 16;
  return *(const v16h*)P;
}
// C/D: lane col = lane&15, row = r + (lane<16 ? 0 : 8)

// ---------------- prep kernels ----------------
// pack W (K x Nout, fp32, row-major) -> f16 WMMA B-fragment order
__global__ void packB_kernel(const float* W, _Float16* P, int K, int Nout) {
  int idx = blockIdx.x * 256 + threadIdx.x;
  if (idx >= K * Nout) return;
  int e    = idx & 15;
  int lane = (idx >> 4) & 31;
  int tile = idx >> 9;                 // 512 packed elements per 32x16 tile
  int ktiles = K >> 5;
  int nt = tile / ktiles, kt = tile - nt * ktiles;
  int col = nt * 16 + (lane & 15);
  int k   = kt * 32 + ((lane < 16) ? 0 : 8) + ((e < 8) ? e : e + 8);
  P[idx] = (_Float16)W[(size_t)k * Nout + col];
}

// emb = gelu(t_embed) @ W + b   (one output element per thread)
__global__ void film_kernel(const float* t, const float* W, const float* bias,
                            float* out, int dout) {
  int idx = blockIdx.x * 256 + threadIdx.x;
  if (idx >= cBS * dout) return;
  int b = idx / dout, o = idx - b * dout;
  float s = bias[o];
  for (int k = 0; k < cDT; ++k) {
    float x = t[b * cDT + k];
    float gx = 0.5f * x * (1.f + tanhf(0.7978845608028654f * (x + 0.044715f * x * x * x)));
    s = fmaf(gx, W[(size_t)k * dout + o], s);
  }
  out[idx] = s;
}

// ---------------- kernel 2: film(X, xtn1) -> Q,K,V ----------------
__global__ __launch_bounds__(256) void qkv_kernel(KP p) {
  __shared__ float    sX[16 * cDX];
  __shared__ _Float16 aXn[16 * cDX];
  __shared__ float    rowm[16];
  int tid = threadIdx.x, lane = tid & 31, wave = tid >> 5;
  int b = blockIdx.x / (cN / 16), it = blockIdx.x % (cN / 16), i0 = it * 16;

  for (int x = tid; x < 16 * cDX; x += 256)
    sX[x] = p.X[((size_t)b * cN + i0 + (x >> 8)) * cDX + (x & 255)];
  if (tid < 16) rowm[tid] = (float)p.M[b * cN + i0 + tid];
  __syncthreads();

  // film xtn1: one wave per 2 rows, LN over 256
#pragma unroll
  for (int rr = 0; rr < 2; ++rr) {
    int row = wave * 2 + rr;
    const float* rp = &sX[row * cDX];
    float s = 0.f, s2 = 0.f;
    for (int c = lane; c < cDX; c += 32) { float v = rp[c]; s += v; s2 += v * v; }
    for (int m = 16; m >= 1; m >>= 1) { s += __shfl_xor(s, m, 32); s2 += __shfl_xor(s2, m, 32); }
    float mu = s * (1.f / cDX), rstd = rsqrtf(s2 * (1.f / cDX) - mu * mu + cEPS);
    for (int c = lane; c < cDX; c += 32) {
      float xn = (rp[c] - mu) * rstd * (1.f + p.fX1[b * 2 * cDX + c]) + p.fX1[b * 2 * cDX + cDX + c];
      aXn[row * cDX + c] = (_Float16)xn;
    }
  }
  __syncthreads();

  const _Float16* Ws[3] = { p.WqH, p.WkH, p.WvH };
  const float*    Bs[3] = { p.qB,  p.kB,  p.vB };
  float*          Os[3] = { p.Qb,  p.Kb,  p.Vb };
  int col = lane & 15, rb = (lane < 16) ? 0 : 8;
  float rm[8];
#pragma unroll
  for (int r = 0; r < 8; ++r) rm[r] = rowm[rb + r];
#pragma unroll
  for (int mtx = 0; mtx < 3; ++mtx) {
#pragma unroll
    for (int q = 0; q < 2; ++q) {
      int ft = wave + q * 8;
      v8f acc = {};
#pragma unroll
      for (int ks = 0; ks < 8; ++ks)
        acc = wmma_f16(ldA(aXn + ks * 32, cDX), ldBp(Ws[mtx], 8, ft, ks), acc);
      int f = ft * 16 + col;
      float bias = Bs[mtx][f];
#pragma unroll
      for (int r = 0; r < 8; ++r)
        Os[mtx][((size_t)b * cN + i0 + rb + r) * cDX + f] = (acc[r] + bias) * rm[r];
    }
  }
}

// ---------------- kernel 3: fused edge pipeline + online softmax ----------------
__global__ __launch_bounds__(256) void edge_kernel(KP p) {
  __shared__ float    sQ[cDX];
  __shared__ float    sK[16 * cDX];
  __shared__ float    sE[16 * cDE];
  __shared__ float    sMid[16 * cDE];
  __shared__ float    sNE[16 * cDE];
  __shared__ float    sY[16 * cDX];
  __shared__ _Float16 aEn[16 * cDE];
  __shared__ _Float16 aY[16 * cDX];
  __shared__ _Float16 aE2[16 * cDE];
  __shared__ _Float16 aH[16 * cFFE];
  __shared__ float    sMj[16];

  int tid = threadIdx.x, lane = tid & 31, wave = tid >> 5;
  int b = blockIdx.x / cN, i = blockIdx.x % cN;
  int g = tid >> 4, gl = tid & 15;              // 16-thread groups, one E-row each
  int col = lane & 15, rbase = (lane < 16) ? 0 : 8;

  float mi = (float)p.M[b * cN + i];
  sQ[tid] = p.Qb[((size_t)b * cN + i) * cDX + tid];
  __syncthreads();

  // ---- hoist all per-thread loop-invariant constants ----
  float qvv[2], bmv[2], bav[2];                 // E1/E2 phase: f = (wave*2+q)*16+col
#pragma unroll
  for (int q = 0; q < 2; ++q) {
    int f = (wave * 2 + q) * 16 + col;
    qvv[q] = sQ[f]; bmv[q] = p.emB[f]; bav[q] = p.eaB[f];
  }
  float beo = p.eoB[(wave & 3) * 16 + col];     // e_out bias (waves 0..3)
  float bf1 = p.lE1B[wave * 16 + col];          // FFN1 bias
  float bf2 = p.lE2B[(wave & 3) * 16 + col];    // FFN2 bias (waves 0..3)
  float g1v[4], b1v[4], g2v[4], b2v[4], s1v[4], h1v[4], s2v[4], h2v[4];
#pragma unroll
  for (int k = 0; k < 4; ++k) {
    int c = gl + 16 * k;
    g1v[k] = p.nE1g[c]; b1v[k] = p.nE1b[c];
    g2v[k] = p.nE2g[c]; b2v[k] = p.nE2b[c];
    s1v[k] = 1.f + p.fE1[b * 2 * cDE + c]; h1v[k] = p.fE1[b * 2 * cDE + cDE + c];
    s2v[k] = 1.f + p.fE2[b * 2 * cDE + c]; h2v[k] = p.fE2[b * 2 * cDE + cDE + c];
  }

  float Mf = -__builtin_inff(), Lf = 0.f, Af = 0.f;   // online softmax state, f = tid
  const float invs = 0.17677669529663687f;            // 1/sqrt(DF=32)

  for (int jt = 0; jt < cN / 16; ++jt) {
    int j0 = jt * 16;
    __syncthreads();
#if HAS_ASYNC_LDS
    {
      int x = tid * 4;  // 4 floats = 16B per thread for the 16x64 E tile
      const float* gE = &p.E[(((size_t)b * cN + i) * cN + j0 + (x >> 6)) * cDE + (x & 63)];
      ASYNC_B128(gE, &sE[x]);
#pragma unroll
      for (int q = 0; q < 4; ++q) {
        int y = tid * 4 + q * 1024;  // 16x256 K tile
        const float* gK = &p.Kb[((size_t)b * cN + j0 + (y >> 8)) * cDX + (y & 255)];
        ASYNC_B128(gK, &sK[y]);
      }
      WAIT_ASYNC();
    }
#else
    for (int x = tid; x < 16 * cDE; x += 256)
      sE[x] = p.E[(((size_t)b * cN + i) * cN + j0 + (x >> 6)) * cDE + (x & 63)];
    for (int x = tid; x < 16 * cDX; x += 256)
      sK[x] = p.Kb[((size_t)b * cN + j0 + (x >> 8)) * cDX + (x & 255)];
#endif
    if (tid < 16) sMj[tid] = (float)p.M[b * cN + j0 + tid];
    __syncthreads();

    // ---- film(E, etn1): LN over 64 per row
    {
      const float* rp = &sE[g * cDE];
      float s = 0.f, s2 = 0.f;
#pragma unroll
      for (int k = 0; k < 4; ++k) { float v = rp[gl + 16 * k]; s += v; s2 += v * v; }
      for (int m = 8; m >= 1; m >>= 1) { s += __shfl_xor(s, m, 32); s2 += __shfl_xor(s2, m, 32); }
      float mu = s * (1.f / cDE), rstd = rsqrtf(s2 * (1.f / cDE) - mu * mu + cEPS);
#pragma unroll
      for (int k = 0; k < 4; ++k) {
        int c = gl + 16 * k;
        aEn[g * cDE + c] = (_Float16)((rp[c] - mu) * rstd * s1v[k] + h1v[k]);
      }
    }
    __syncthreads();

    // ---- E1/E2 GEMMs (64 -> 256 each) and Y
#pragma unroll
    for (int q = 0; q < 2; ++q) {
      int ft = wave * 2 + q;
      v8f a1 = {}, a2 = {};
#pragma unroll
      for (int ks = 0; ks < 2; ++ks) {
        v16h a = ldA(aEn + ks * 32, cDE);
        a1 = wmma_f16(a, ldBp(p.WemH, 2, ft, ks), a1);
        a2 = wmma_f16(a, ldBp(p.WeaH, 2, ft, ks), a2);
      }
      int f = ft * 16 + col;
      float qv = qvv[q], bm = bmv[q], ba = bav[q];
#pragma unroll
      for (int r = 0; r < 8; ++r) {
        int row = rbase + r;
        float mk = mi * sMj[row];
        float e1 = (a1[r] + bm) * mk;
        float e2 = (a2[r] + ba) * mk;
        float y = qv * sK[row * cDX + f] * invs * (e1 + 1.f) + e2;
        sY[row * cDX + f] = y;
        aY[row * cDX + f] = (_Float16)y;
      }
    }
    __syncthreads();

    // ---- online softmax + wV accumulation (thread tid owns feature f = tid)
    for (int r = 0; r < 16; ++r) {
      if (sMj[r] > 0.f) {
        float y  = sY[r * cDX + tid];
        float nM = fmaxf(Mf, y);
        float cr = __expf(Mf - nM);
        float e  = __expf(y - nM);
        Lf = Lf * cr + e;
        Af = Af * cr + e * p.Vb[((size_t)b * cN + j0 + r) * cDX + tid];
        Mf = nM;
      }
    }

    // ---- newE = Y @ We_out (256 -> 64), masked (waves 0..3)
    if (wave < 4) {
      v8f acc = {};
#pragma unroll
      for (int ks = 0; ks < 8; ++ks)
        acc = wmma_f16(ldA(aY + ks * 32, cDX), ldBp(p.WeoH, 8, wave, ks), acc);
      int c = wave * 16 + col;
#pragma unroll
      for (int r = 0; r < 8; ++r) {
        int row = rbase + r;
        sNE[row * cDE + c] = (acc[r] + beo) * mi * sMj[row];
      }
    }
    __syncthreads();

    // ---- ln_aff(normE1) + residual -> sMid ; film(etn2) -> aE2
    {
      const float* rp = &sNE[g * cDE];
      float s = 0.f, s2 = 0.f;
#pragma unroll
      for (int k = 0; k < 4; ++k) { float v = rp[gl + 16 * k]; s += v; s2 += v * v; }
      for (int m = 8; m >= 1; m >>= 1) { s += __shfl_xor(s, m, 32); s2 += __shfl_xor(s2, m, 32); }
      float mu = s * (1.f / cDE), rstd = rsqrtf(s2 * (1.f / cDE) - mu * mu + cEPS);
#pragma unroll
      for (int k = 0; k < 4; ++k) {
        int c = gl + 16 * k;
        sMid[g * cDE + c] = (rp[c] - mu) * rstd * g1v[k] + b1v[k] + sE[g * cDE + c];
      }
      __builtin_amdgcn_wave_barrier();
      const float* rp2 = &sMid[g * cDE];
      s = 0.f; s2 = 0.f;
#pragma unroll
      for (int k = 0; k < 4; ++k) { float v = rp2[gl + 16 * k]; s += v; s2 += v * v; }
      for (int m = 8; m >= 1; m >>= 1) { s += __shfl_xor(s, m, 32); s2 += __shfl_xor(s2, m, 32); }
      mu = s * (1.f / cDE); rstd = rsqrtf(s2 * (1.f / cDE) - mu * mu + cEPS);
#pragma unroll
      for (int k = 0; k < 4; ++k) {
        int c = gl + 16 * k;
        aE2[g * cDE + c] = (_Float16)((rp2[c] - mu) * rstd * s2v[k] + h2v[k]);
      }
    }
    __syncthreads();

    // ---- FFN1: relu(En2 @ WE1 + b)  (64 -> 128)
    {
      v8f acc = {};
#pragma unroll
      for (int ks = 0; ks < 2; ++ks)
        acc = wmma_f16(ldA(aE2 + ks * 32, cDE), ldBp(p.WE1H, 2, wave, ks), acc);
      int f = wave * 16 + col;
#pragma unroll
      for (int r = 0; r < 8; ++r)
        aH[(rbase + r) * cFFE + f] = (_Float16)fmaxf(acc[r] + bf1, 0.f);
    }
    __syncthreads();

    // ---- FFN2: H @ WE2 + b  (128 -> 64)  (waves 0..3)
    if (wave < 4) {
      v8f acc = {};
#pragma unroll
      for (int ks = 0; ks < 4; ++ks)
        acc = wmma_f16(ldA(aH + ks * 32, cFFE), ldBp(p.WE2H, 4, wave, ks), acc);
      int c = wave * 16 + col;
#pragma unroll
      for (int r = 0; r < 8; ++r)
        sNE[(rbase + r) * cDE + c] = acc[r] + bf2;
    }
    __syncthreads();

    // ---- final ln_aff(normE2) + sMid -> E out
    {
      const float* rp = &sNE[g * cDE];
      float s = 0.f, s2 = 0.f;
#pragma unroll
      for (int k = 0; k < 4; ++k) { float v = rp[gl + 16 * k]; s += v; s2 += v * v; }
      for (int m = 8; m >= 1; m >>= 1) { s += __shfl_xor(s, m, 32); s2 += __shfl_xor(s2, m, 32); }
      float mu = s * (1.f / cDE), rstd = rsqrtf(s2 * (1.f / cDE) - mu * mu + cEPS);
#pragma unroll
      for (int k = 0; k < 4; ++k) {
        int c = gl + 16 * k;
        p.Eout[(((size_t)b * cN + i) * cN + j0 + g) * cDE + c] =
            (rp[c] - mu) * rstd * g2v[k] + b2v[k] + sMid[g * cDE + c];
      }
    }
  }
  p.wVb[((size_t)b * cN + i) * cDX + tid] = Af / Lf;
}

// ---------------- kernel 4: X tail (x_out GEMM, LNs, FFN 256->1024->256) ----------------
__global__ __launch_bounds__(256) void xtail_kernel(KP p) {
  __shared__ _Float16 aA[16 * cDX];
  __shared__ float    sMid[16 * cDX];
  __shared__ float    sT[16 * cDX];
  __shared__ _Float16 aHh[16 * (cFFX / 2)];
  __shared__ float    rowm[16];
  int tid = threadIdx.x, lane = tid & 31, wave = tid >> 5;
  int b = blockIdx.x / (cN / 16), it = blockIdx.x % (cN / 16), i0 = it * 16;
  int col = lane & 15, rb = (lane < 16) ? 0 : 8;

  for (int x = tid; x < 16 * cDX; x += 256)
    aA[x] = (_Float16)p.wVb[((size_t)b * cN + i0 + (x >> 8)) * cDX + (x & 255)];
  if (tid < 16) rowm[tid] = (float)p.M[b * cN + i0 + tid];
  __syncthreads();

  // newX = wV @ Wx_out + b, masked
#pragma unroll
  for (int q = 0; q < 2; ++q) {
    int ft = wave + q * 8;
    v8f acc = {};
#pragma unroll
    for (int ks = 0; ks < 8; ++ks)
      acc = wmma_f16(ldA(aA + ks * 32, cDX), ldBp(p.WxoH, 8, ft, ks), acc);
    int f = ft * 16 + col;
    float bias = p.xoB[f];
#pragma unroll
    for (int r = 0; r < 8; ++r)
      sT[(rb + r) * cDX + f] = (acc[r] + bias) * rowm[rb + r];
  }
  __syncthreads();

  // ln_aff(normX1) + X residual -> sMid ; film(xtn2) -> aA (reuse)
#pragma unroll
  for (int rr = 0; rr < 2; ++rr) {
    int row = wave * 2 + rr;
    const float* rp = &sT[row * cDX];
    float s = 0.f, s2 = 0.f;
    for (int c = lane; c < cDX; c += 32) { float v = rp[c]; s += v; s2 += v * v; }
    for (int m = 16; m >= 1; m >>= 1) { s += __shfl_xor(s, m, 32); s2 += __shfl_xor(s2, m, 32); }
    float mu = s * (1.f / cDX), rstd = rsqrtf(s2 * (1.f / cDX) - mu * mu + cEPS);
    for (int c = lane; c < cDX; c += 32)
      sMid[row * cDX + c] = (rp[c] - mu) * rstd * p.nX1g[c] + p.nX1b[c] +
                            p.X[((size_t)b * cN + i0 + row) * cDX + c];
    __builtin_amdgcn_wave_barrier();
    const float* rp2 = &sMid[row * cDX];
    s = 0.f; s2 = 0.f;
    for (int c = lane; c < cDX; c += 32) { float v = rp2[c]; s += v; s2 += v * v; }
    for (int m = 16; m >= 1; m >>= 1) { s += __shfl_xor(s, m, 32); s2 += __shfl_xor(s2, m, 32); }
    mu = s * (1.f / cDX); rstd = rsqrtf(s2 * (1.f / cDX) - mu * mu + cEPS);
    for (int c = lane; c < cDX; c += 32)
      aA[row * cDX + c] = (_Float16)((rp2[c] - mu) * rstd * (1.f + p.fX2[b * 2 * cDX + c]) +
                                     p.fX2[b * 2 * cDX + cDX + c]);
  }
  __syncthreads();

  // FFN 256 -> 1024 -> 256, processed in two K-halves to bound LDS
  v8f accO0 = {}, accO1 = {};
  for (int half = 0; half < 2; ++half) {
#pragma unroll
    for (int q = 0; q < 4; ++q) {
      int lft = wave * 4 + q;               // local f-tile 0..31 (512 cols)
      v8f acc = {};
#pragma unroll
      for (int ks = 0; ks < 8; ++ks)
        acc = wmma_f16(ldA(aA + ks * 32, cDX), ldBp(p.WX1H, 8, half * 32 + lft, ks), acc);
      int lf = lft * 16 + col;
      float bias = p.lX1B[half * 512 + lf];
#pragma unroll
      for (int r = 0; r < 8; ++r)
        aHh[(rb + r) * 512 + lf] = (_Float16)fmaxf(acc[r] + bias, 0.f);
    }
    __syncthreads();
#pragma unroll
    for (int q = 0; q < 2; ++q) {
      int ft = wave + q * 8;
      v8f acc = (q == 0) ? accO0 : accO1;
#pragma unroll
      for (int ks = 0; ks < 16; ++ks)
        acc = wmma_f16(ldA(aHh + ks * 32, 512), ldBp(p.WX2H, 32, ft, half * 16 + ks), acc);
      if (q == 0) accO0 = acc; else accO1 = acc;
    }
    __syncthreads();
  }
#pragma unroll
  for (int q = 0; q < 2; ++q) {
    int ft = wave + q * 8, f = ft * 16 + col;
    v8f acc = (q == 0) ? accO0 : accO1;
    float bias = p.lX2B[f];
#pragma unroll
    for (int r = 0; r < 8; ++r)
      sT[(rb + r) * cDX + f] = acc[r] + bias;
  }
  __syncthreads();

  // final ln_aff(normX2) + sMid -> X out
#pragma unroll
  for (int rr = 0; rr < 2; ++rr) {
    int row = wave * 2 + rr;
    const float* rp = &sT[row * cDX];
    float s = 0.f, s2 = 0.f;
    for (int c = lane; c < cDX; c += 32) { float v = rp[c]; s += v; s2 += v * v; }
    for (int m = 16; m >= 1; m >>= 1) { s += __shfl_xor(s, m, 32); s2 += __shfl_xor(s2, m, 32); }
    float mu = s * (1.f / cDX), rstd = rsqrtf(s2 * (1.f / cDX) - mu * mu + cEPS);
    for (int c = lane; c < cDX; c += 32)
      p.Xout[((size_t)b * cN + i0 + row) * cDX + c] =
          (rp[c] - mu) * rstd * p.nX2g[c] + p.nX2b[c] + sMid[row * cDX + c];
  }
}

// ---------------- kernel 5: y path (tiny, fp32 scalar) ----------------
__global__ __launch_bounds__(256) void y_kernel(KP p) {
  __shared__ float y1[cDY], hh[cFFY], uu[cDY], st[2];
  int t = threadIdx.x, lane = t & 31;
  for (int b = 0; b < cBS; ++b) {
    __syncthreads();
    if (t < 32) {
      float s = 0.f, s2 = 0.f;
      for (int c = lane; c < cDY; c += 32) { float v = p.Y[b * cDY + c]; s += v; s2 += v * v; }
      for (int m = 16; m >= 1; m >>= 1) { s += __shfl_xor(s, m, 32); s2 += __shfl_xor(s2, m, 32); }
      if (lane == 0) { float mu = s / cDY; st[0] = mu; st[1] = rsqrtf(s2 / cDY - mu * mu + cEPS); }
    }
    __syncthreads();
    if (t < cDY) { float v = p.Y[b * cDY + t]; y1[t] = (v - st[0]) * st[1] * p.ny1g[t] + p.ny1b[t] + v; }
    __syncthreads();
    for (int o = t; o < cFFY; o += 256) {
      float s = p.ly1B[o];
      for (int k = 0; k < cDY; ++k) s = fmaf(y1[k], p.ly1W[(size_t)k * cFFY + o], s);
      hh[o] = fmaxf(s, 0.f);
    }
    __syncthreads();
    if (t < cDY) {
      float s = p.ly2B[t];
      for (int k = 0; k < cFFY; ++k) s = fmaf(hh[k], p.ly2W[(size_t)k * cDY + t], s);
      uu[t] = s;
    }
    __syncthreads();
    if (t < 32) {
      float s = 0.f, s2 = 0.f;
      for (int c = lane; c < cDY; c += 32) { float v = uu[c]; s += v; s2 += v * v; }
      for (int m = 16; m >= 1; m >>= 1) { s += __shfl_xor(s, m, 32); s2 += __shfl_xor(s2, m, 32); }
      if (lane == 0) { float mu = s / cDY; st[0] = mu; st[1] = rsqrtf(s2 / cDY - mu * mu + cEPS); }
    }
    __syncthreads();
    if (t < cDY) p.yout[b * cDY + t] = (uu[t] - st[0]) * st[1] * p.ny2g[t] + p.ny2b[t] + y1[t];
  }
}

// ---------------- host launcher ----------------
extern "C" void kernel_launch(void* const* d_in, const int* in_sizes, int n_in,
                              void* d_out, int out_size, void* d_ws, size_t ws_size,
                              hipStream_t stream) {
  (void)in_sizes; (void)out_size; (void)ws_size;
  // params in dict-insertion order; linear: {w,b}; norm: {g,b}
  static const int leafSize[46] = {
    65536,256, 65536,256, 65536,256,            // q, k, v
    16384,256, 16384,256,                       // e_mul, e_add
    65536,256, 16384,64,                        // x_out, e_out
    131072,512, 131072,512,                     // xtn1, xtn2
    32768,128, 32768,128,                       // etn1, etn2
    262144,1024, 262144,256,                    // linX1, linX2
    8192,128, 8192,64,                          // linE1, linE2
    65536,512, 65536,128,                       // liny1, liny2
    256,256, 256,256, 64,64, 64,64, 128,128, 128,128 }; // norms
  const float* leaf[46];
  if (n_in >= 51) {
    for (int i = 0; i < 46; ++i) leaf[i] = (const float*)d_in[4 + i];
  } else {
    const float* base = (const float*)d_in[4];
    size_t off = 0;
    for (int i = 0; i < 46; ++i) { leaf[i] = base + off; off += (size_t)leafSize[i]; }
  }
  const int* mask = (const int*)d_in[n_in - 1];

  // workspace carve-out
  char* w = (char*)d_ws;
  auto take = [&](size_t bytes) -> char* {
    char* r = w; w += (bytes + 255) & ~(size_t)255; return r;
  };
  _Float16* WqH  = (_Float16*)take(65536 * 2);
  _Float16* WkH  = (_Float16*)take(65536 * 2);
  _Float16* WvH  = (_Float16*)take(65536 * 2);
  _Float16* WxoH = (_Float16*)take(65536 * 2);
  _Float16* WemH = (_Float16*)take(16384 * 2);
  _Float16* WeaH = (_Float16*)take(16384 * 2);
  _Float16* WeoH = (_Float16*)take(16384 * 2);
  _Float16* WE1H = (_Float16*)take(8192 * 2);
  _Float16* WE2H = (_Float16*)take(8192 * 2);
  _Float16* WX1H = (_Float16*)take(262144 * 2);
  _Float16* WX2H = (_Float16*)take(262144 * 2);
  float* fX1 = (float*)take((size_t)cBS * 2 * cDX * 4);
  float* fX2 = (float*)take((size_t)cBS * 2 * cDX * 4);
  float* fE1 = (float*)take((size_t)cBS * 2 * cDE * 4);
  float* fE2 = (float*)take((size_t)cBS * 2 * cDE * 4);
  float* Qb  = (float*)take((size_t)cBS * cN * cDX * 4);
  float* Kb  = (float*)take((size_t)cBS * cN * cDX * 4);
  float* Vb  = (float*)take((size_t)cBS * cN * cDX * 4);
  float* wVb = (float*)take((size_t)cBS * cN * cDX * 4);

  KP p{};
  p.X = (const float*)d_in[0]; p.E = (const float*)d_in[1];
  p.Y = (const float*)d_in[2]; p.T = (const float*)d_in[3]; p.M = mask;
  p.qB = leaf[1]; p.kB = leaf[3]; p.vB = leaf[5];
  p.emB = leaf[7]; p.eaB = leaf[9]; p.xoB = leaf[11]; p.eoB = leaf[13];
  p.lX1B = leaf[23]; p.lX2B = leaf[25]; p.lE1B = leaf[27]; p.lE2B = leaf[29];
  p.ly1W = leaf[30]; p.ly1B = leaf[31]; p.ly2W = leaf[32]; p.ly2B = leaf[33];
  p.nX1g = leaf[34]; p.nX1b = leaf[35]; p.nX2g = leaf[36]; p.nX2b = leaf[37];
  p.nE1g = leaf[38]; p.nE1b = leaf[39]; p.nE2g = leaf[40]; p.nE2b = leaf[41];
  p.ny1g = leaf[42]; p.ny1b = leaf[43]; p.ny2g = leaf[44]; p.ny2b = leaf[45];
  p.WqH = WqH; p.WkH = WkH; p.WvH = WvH; p.WxoH = WxoH;
  p.WemH = WemH; p.WeaH = WeaH; p.WeoH = WeoH;
  p.WE1H = WE1H; p.WE2H = WE2H; p.WX1H = WX1H; p.WX2H = WX2H;
  p.fX1 = fX1; p.fX2 = fX2; p.fE1 = fE1; p.fE2 = fE2;
  p.Qb = Qb; p.Kb = Kb; p.Vb = Vb; p.wVb = wVb;
  p.Xout = (float*)d_out;
  p.Eout = (float*)d_out + (size_t)cBS * cN * cDX;
  p.yout = p.Eout + (size_t)cBS * cN * cN * cDE;

  auto pack = [&](const float* s, _Float16* d, int K, int Nout) {
    int n = K * Nout;
    packB_kernel<<<dim3((n + 255) / 256), dim3(256), 0, stream>>>(s, d, K, Nout);
  };
  pack(leaf[0],  WqH,  256, 256);
  pack(leaf[2],  WkH,  256, 256);
  pack(leaf[4],  WvH,  256, 256);
  pack(leaf[10], WxoH, 256, 256);
  pack(leaf[6],  WemH, 64,  256);
  pack(leaf[8],  WeaH, 64,  256);
  pack(leaf[12], WeoH, 256, 64);
  pack(leaf[26], WE1H, 64,  128);
  pack(leaf[28], WE2H, 128, 64);
  pack(leaf[22], WX1H, 256, 1024);
  pack(leaf[24], WX2H, 1024, 256);

  film_kernel<<<dim3((cBS * 2 * cDX + 255) / 256), dim3(256), 0, stream>>>(p.T, leaf[14], leaf[15], fX1, 2 * cDX);
  film_kernel<<<dim3((cBS * 2 * cDX + 255) / 256), dim3(256), 0, stream>>>(p.T, leaf[16], leaf[17], fX2, 2 * cDX);
  film_kernel<<<dim3((cBS * 2 * cDE + 255) / 256), dim3(256), 0, stream>>>(p.T, leaf[18], leaf[19], fE1, 2 * cDE);
  film_kernel<<<dim3((cBS * 2 * cDE + 255) / 256), dim3(256), 0, stream>>>(p.T, leaf[20], leaf[21], fE2, 2 * cDE);

  qkv_kernel  <<<dim3(cBS * (cN / 16)), dim3(256), 0, stream>>>(p);
  edge_kernel <<<dim3(cBS * cN),        dim3(256), 0, stream>>>(p);
  xtail_kernel<<<dim3(cBS * (cN / 16)), dim3(256), 0, stream>>>(p);
  y_kernel    <<<dim3(1),               dim3(256), 0, stream>>>(p);
}
